// CustomMultiHeadSelfAttention_81905026335125
// MI455X (gfx1250) — matrix operand order repaired and checked by reading, not compiled, not run on hardware
//
#include <hip/hip_runtime.h>

// ---------------- problem constants ----------------
#define Bb 2
#define Ss 2048
#define Ee 1024
#define Hh 16
#define Dd 64
#define N3 3072   // 3*E

typedef __bf16 bf16;
typedef __attribute__((ext_vector_type(2)))  bf16  v2bf;
typedef __attribute__((ext_vector_type(8)))  bf16  v8bf;
typedef __attribute__((ext_vector_type(16))) bf16  v16bf;
typedef __attribute__((ext_vector_type(8)))  float v8f;
typedef __attribute__((ext_vector_type(4)))  float v4f;

__device__ __forceinline__ v16bf frag_cat(v8bf lo, v8bf hi) {
  return __builtin_shufflevector(lo, hi, 0,1,2,3,4,5,6,7,8,9,10,11,12,13,14,15);
}
__device__ __forceinline__ v8f wmma_bf16(v16bf a, v16bf b, v8f c) {
  // D = A(16x32) * B(32x16) + C, fp32 accumulate
  return __builtin_amdgcn_wmma_f32_16x16x32_bf16(false, a, false, b, (short)0, c,
                                                 false, false);
}

// =====================================================================
// Shared GEMM core: 64x128 block tile, 8 waves, K-step 32.
// Wave w: row-tile tm = w&3 (A reused across 4 col-tiles), cols (w>>2)*4+t.
// All fragment ds_loads are issued BEFORE the WMMA chain so the loads
// retire under partial s_wait_dscnt while earlier WMMAs execute.
// =====================================================================

// ---- Kernel 1: qkv = x @ W_attn + b_attn; scatter Q,K [B,H,S,D], V^T [B,H,D,S]
__global__ __launch_bounds__(256)
void qkv_gemm_kernel(const float* __restrict__ x, const float* __restrict__ W,
                     const float* __restrict__ bias,
                     bf16* __restrict__ Qb, bf16* __restrict__ Kb,
                     bf16* __restrict__ Vt)
{
  __shared__ bf16 As [64][40];    // [row][k]   80B stride
  __shared__ bf16 Bts[128][40];   // [n][k]     transposed W
  const int tid  = threadIdx.x;
  const int lane = tid & 31, wave = tid >> 5;
  const int g = lane >> 4, m = lane & 15;
  const int bm = blockIdx.y, bn = blockIdx.x;
  const int tm = wave & 3;            // A row sub-tile
  const int cg = wave >> 2;           // column group: 4 tiles each
  const int arow = tid >> 2, acs = (tid & 3) * 8;   // A staging: 4 thr/row
  const int kp  = (tid >> 4) * 2;     // W staging: even k row pair
  const int nsB = (tid & 15) * 8;     // 8 contiguous columns

  v8f zero = {};
  v8f c[4] = {zero, zero, zero, zero};

  for (int k0 = 0; k0 < Ee; k0 += 32) {
    // ---- stage A (coalesced fp32 -> bf16) ----
    {
      const float* src = x + (size_t)(bm * 64 + arow) * Ee + k0 + acs;
      v4f a0 = *(const v4f*)(src);
      v4f a1 = *(const v4f*)(src + 4);
#pragma unroll
      for (int i = 0; i < 4; ++i) {
        As[arow][acs + i]     = (bf16)a0[i];
        As[arow][acs + 4 + i] = (bf16)a1[i];
      }
    }
    // ---- stage W: coalesced reads of 2 k-rows, packed bf16-pair scatter ----
    {
      const float* w0 = W + (size_t)(k0 + kp) * N3 + bn * 128 + nsB;
      v4f r0a = *(const v4f*)(w0),      r0b = *(const v4f*)(w0 + 4);
      v4f r1a = *(const v4f*)(w0 + N3), r1b = *(const v4f*)(w0 + N3 + 4);
#pragma unroll
      for (int i = 0; i < 4; ++i) {
        v2bf p; p[0] = (bf16)r0a[i]; p[1] = (bf16)r1a[i];
        *(v2bf*)&Bts[nsB + i][kp] = p;          // ds_store_b32 / 2addr
        v2bf q; q[0] = (bf16)r0b[i]; q[1] = (bf16)r1b[i];
        *(v2bf*)&Bts[nsB + 4 + i][kp] = q;
      }
    }
    __syncthreads();

    // A fragment 16x32: e -> k = g*16 + e
    v16bf a = frag_cat(*(const v8bf*)&As[tm * 16 + m][g * 16],
                       *(const v8bf*)&As[tm * 16 + m][g * 16 + 8]);
    // hoist ALL B fragments (32x16: lane n = m, e -> k = g*16 + e) ...
    v16bf bf[4];
#pragma unroll
    for (int t = 0; t < 4; ++t)
      bf[t] = frag_cat(*(const v8bf*)&Bts[(cg * 4 + t) * 16 + m][g * 16],
                       *(const v8bf*)&Bts[(cg * 4 + t) * 16 + m][g * 16 + 8]);
    // ... then run the WMMA chain with loads draining underneath
#pragma unroll
    for (int t = 0; t < 4; ++t) c[t] = wmma_bf16(a, bf[t], c[t]);
    __syncthreads();
  }

  // ---- epilogue: bias + scatter into Q / K / V^T ----
#pragma unroll
  for (int t = 0; t < 4; ++t) {
    int n = bn * 128 + (cg * 4 + t) * 16 + m;
    float bv = bias[n];
#pragma unroll
    for (int r = 0; r < 8; ++r) {
      int row = bm * 64 + tm * 16 + r + 8 * g;   // C layout: m = r + 8g
      int b = row >> 11, s = row & (Ss - 1);
      float v = c[t][r] + bv;
      if (n < Ee) {
        int h = n >> 6, d = n & 63;
        Qb[(((size_t)(b * Hh + h) * Ss) + s) * Dd + d] = (bf16)v;
      } else if (n < 2 * Ee) {
        int n2 = n - Ee, h = n2 >> 6, d = n2 & 63;
        Kb[(((size_t)(b * Hh + h) * Ss) + s) * Dd + d] = (bf16)v;
      } else {
        int n2 = n - 2 * Ee, h = n2 >> 6, d = n2 & 63;
        Vt[(((size_t)(b * Hh + h) * Dd) + d) * Ss + s] = (bf16)v;
      }
    }
  }
}

// =====================================================================
// Kernel 2: flash attention. Wave owns 16 query rows; 64 keys per step:
// 8 WMMA (Q.K^T) + base-2 online softmax + 8 WMMA (P.V).
// =====================================================================
__global__ __launch_bounds__(256)
void attn_kernel(const bf16* __restrict__ Qb, const bf16* __restrict__ Kb,
                 const bf16* __restrict__ Vt, bf16* __restrict__ Y)
{
  __shared__ bf16 Ps[8][16][72];   // per-wave P bounce 16x64 (+8 pad)
  const int tid  = threadIdx.x;
  const int lane = tid & 31, wave = tid >> 5;
  const int g = lane >> 4, m = lane & 15;
  const int h = blockIdx.y, bb = blockIdx.z;
  const int bh = bb * Hh + h;
  const int qbase = blockIdx.x * 128 + wave * 16;
  const float SC = 0.125f * 1.44269504088896f;   // 1/sqrt(D) * log2(e)

  const bf16* Qh = Qb + (size_t)bh * Ss * Dd;
  const bf16* Kh = Kb + (size_t)bh * Ss * Dd;
  const bf16* Vh = Vt + (size_t)bh * Dd * Ss;

  // Q A-fragments, d in [0,32) / [32,64), live in registers for whole loop
  const bf16* qrow = Qh + (size_t)(qbase + m) * Dd;
  v16bf q0 = frag_cat(*(const v8bf*)(qrow + g * 16),
                      *(const v8bf*)(qrow + g * 16 + 8));
  v16bf q1 = frag_cat(*(const v8bf*)(qrow + 32 + g * 16),
                      *(const v8bf*)(qrow + 40 + g * 16));

  float mrow[8], lrow[8];
  v8f zero = {};
  v8f o[4] = {zero, zero, zero, zero};   // O accumulator 16x64
#pragma unroll
  for (int r = 0; r < 8; ++r) { mrow[r] = -3.0e38f; lrow[r] = 0.0f; }

  for (int j0 = 0; j0 < Ss; j0 += 64) {
    // ---- hoist all 8 K B-fragments (one vmem clause), then 8 WMMA ----
    v16bf kf[8];
#pragma unroll
    for (int t = 0; t < 4; ++t) {
      const bf16* krow = Kh + (size_t)(j0 + t * 16 + m) * Dd;
      kf[2 * t]     = frag_cat(*(const v8bf*)(krow + g * 16),
                               *(const v8bf*)(krow + g * 16 + 8));
      kf[2 * t + 1] = frag_cat(*(const v8bf*)(krow + 32 + g * 16),
                               *(const v8bf*)(krow + 32 + g * 16 + 8));
    }
    v8f s0 = zero, s1 = zero, s2 = zero, s3 = zero;
    s0 = wmma_bf16(q0, kf[0], s0);  s0 = wmma_bf16(q1, kf[1], s0);
    s1 = wmma_bf16(q0, kf[2], s1);  s1 = wmma_bf16(q1, kf[3], s1);
    s2 = wmma_bf16(q0, kf[4], s2);  s2 = wmma_bf16(q1, kf[5], s2);
    s3 = wmma_bf16(q0, kf[6], s3);  s3 = wmma_bf16(q1, kf[7], s3);

    if (j0 + 64 < Ss)   // gfx1250 global_prefetch_b8 on next K tile
      __builtin_prefetch(Kh + (size_t)(j0 + 64 + m) * Dd, 0, 3);

    // ---- base-2 online softmax; each row = one 16-lane half-group ----
    float alpha[8];
#pragma unroll
    for (int r = 0; r < 8; ++r) {
      float a0 = s0[r] * SC, a1 = s1[r] * SC;
      float a2 = s2[r] * SC, a3 = s3[r] * SC;
      float mx = fmaxf(fmaxf(a0, a1), fmaxf(a2, a3));
      mx = fmaxf(mx, __shfl_xor(mx, 1));
      mx = fmaxf(mx, __shfl_xor(mx, 2));
      mx = fmaxf(mx, __shfl_xor(mx, 4));
      mx = fmaxf(mx, __shfl_xor(mx, 8));
      float mn = fmaxf(mrow[r], mx);
      float al = exp2f(mrow[r] - mn);          // v_exp_f32 directly
      float e0 = exp2f(a0 - mn), e1 = exp2f(a1 - mn);
      float e2 = exp2f(a2 - mn), e3 = exp2f(a3 - mn);
      float sm = (e0 + e1) + (e2 + e3);
      sm += __shfl_xor(sm, 1);
      sm += __shfl_xor(sm, 2);
      sm += __shfl_xor(sm, 4);
      sm += __shfl_xor(sm, 8);
      lrow[r] = lrow[r] * al + sm;
      mrow[r] = mn;
      alpha[r] = al;
      // write P row into per-wave LDS bounce (C-layout -> memory order)
      int mm = r + 8 * g;
      Ps[wave][mm][m]      = (bf16)e0;
      Ps[wave][mm][16 + m] = (bf16)e1;
      Ps[wave][mm][32 + m] = (bf16)e2;
      Ps[wave][mm][48 + m] = (bf16)e3;
    }
#pragma unroll
    for (int t = 0; t < 4; ++t)
#pragma unroll
      for (int r = 0; r < 8; ++r) o[t][r] *= alpha[r];

    asm volatile("s_wait_dscnt 0" ::: "memory");   // wave-local DS RAW fence
    // P as two A-fragments (k = P columns 0..31 / 32..63)
    v16bf pf0 = frag_cat(*(const v8bf*)&Ps[wave][m][g * 16],
                         *(const v8bf*)&Ps[wave][m][g * 16 + 8]);
    v16bf pf1 = frag_cat(*(const v8bf*)&Ps[wave][m][32 + g * 16],
                         *(const v8bf*)&Ps[wave][m][40 + g * 16]);

    // ---- hoist all 8 V B-fragments, then 8 WMMA: O += P @ V ----
    v16bf vf[8];
#pragma unroll
    for (int t = 0; t < 4; ++t) {
      const bf16* vrow = Vh + (size_t)(t * 16 + m) * Ss + j0 + g * 16;
      vf[2 * t]     = frag_cat(*(const v8bf*)(vrow),      *(const v8bf*)(vrow + 8));
      vf[2 * t + 1] = frag_cat(*(const v8bf*)(vrow + 32), *(const v8bf*)(vrow + 40));
    }
#pragma unroll
    for (int t = 0; t < 4; ++t) {
      o[t] = wmma_bf16(pf0, vf[2 * t], o[t]);
      o[t] = wmma_bf16(pf1, vf[2 * t + 1], o[t]);
    }
  }

  // ---- normalize and store Y[b][s][h][d] (bf16) ----
#pragma unroll
  for (int r = 0; r < 8; ++r) {
    float inv = 1.0f / lrow[r];
    int s = qbase + r + 8 * g;
    bf16* yrow = Y + (((size_t)bb * Ss + s) * Hh + h) * Dd;
#pragma unroll
    for (int t = 0; t < 4; ++t) yrow[t * 16 + m] = (bf16)(o[t][r] * inv);
  }
}

// =====================================================================
// Kernel 3: out = Y @ W_proj + b_proj  (bf16 A, fp32 W/out), 64x128 tile
// =====================================================================
__global__ __launch_bounds__(256)
void proj_gemm_kernel(const bf16* __restrict__ Yin, const float* __restrict__ W,
                      const float* __restrict__ bias, float* __restrict__ out)
{
  __shared__ bf16 As [64][40];
  __shared__ bf16 Bts[128][40];
  const int tid  = threadIdx.x;
  const int lane = tid & 31, wave = tid >> 5;
  const int g = lane >> 4, m = lane & 15;
  const int bm = blockIdx.y, bn = blockIdx.x;
  const int tm = wave & 3, cg = wave >> 2;
  const int arow = tid >> 2, acs = (tid & 3) * 8;
  const int kp  = (tid >> 4) * 2;
  const int nsB = (tid & 15) * 8;

  v8f zero = {};
  v8f c[4] = {zero, zero, zero, zero};

  for (int k0 = 0; k0 < Ee; k0 += 32) {
    // A already bf16: straight 16B copy into LDS
    *(v8bf*)&As[arow][acs] =
        *(const v8bf*)(Yin + (size_t)(bm * 64 + arow) * Ee + k0 + acs);
    // W: coalesced fp32 reads of 2 k-rows, packed bf16-pair transposed scatter
    {
      const float* w0 = W + (size_t)(k0 + kp) * Ee + bn * 128 + nsB;
      v4f r0a = *(const v4f*)(w0),      r0b = *(const v4f*)(w0 + 4);
      v4f r1a = *(const v4f*)(w0 + Ee), r1b = *(const v4f*)(w0 + Ee + 4);
#pragma unroll
      for (int i = 0; i < 4; ++i) {
        v2bf p; p[0] = (bf16)r0a[i]; p[1] = (bf16)r1a[i];
        *(v2bf*)&Bts[nsB + i][kp] = p;
        v2bf q; q[0] = (bf16)r0b[i]; q[1] = (bf16)r1b[i];
        *(v2bf*)&Bts[nsB + 4 + i][kp] = q;
      }
    }
    __syncthreads();

    v16bf a = frag_cat(*(const v8bf*)&As[tm * 16 + m][g * 16],
                       *(const v8bf*)&As[tm * 16 + m][g * 16 + 8]);
    v16bf bf[4];
#pragma unroll
    for (int t = 0; t < 4; ++t)
      bf[t] = frag_cat(*(const v8bf*)&Bts[(cg * 4 + t) * 16 + m][g * 16],
                       *(const v8bf*)&Bts[(cg * 4 + t) * 16 + m][g * 16 + 8]);
#pragma unroll
    for (int t = 0; t < 4; ++t) c[t] = wmma_bf16(a, bf[t], c[t]);
    __syncthreads();
  }

#pragma unroll
  for (int t = 0; t < 4; ++t) {
    int n = bn * 128 + (cg * 4 + t) * 16 + m;
    float bv = bias[n];
#pragma unroll
    for (int r = 0; r < 8; ++r) {
      int row = bm * 64 + tm * 16 + r + 8 * g;
      out[(size_t)row * Ee + n] = c[t][r] + bv;
    }
  }
}

// =====================================================================
extern "C" void kernel_launch(void* const* d_in, const int* in_sizes, int n_in,
                              void* d_out, int out_size, void* d_ws, size_t ws_size,
                              hipStream_t stream) {
  const float* x      = (const float*)d_in[0];
  const float* W_attn = (const float*)d_in[1];
  const float* b_attn = (const float*)d_in[2];
  const float* W_proj = (const float*)d_in[3];
  const float* b_proj = (const float*)d_in[4];

  // workspace carve (bf16): Q | K | V^T | Y, 8 MB each
  const size_t perBuf = (size_t)Bb * Hh * Ss * Dd;   // 4 Mi elements
  bf16* Qb = (bf16*)d_ws;
  bf16* Kb = Qb + perBuf;
  bf16* Vt = Kb + perBuf;
  bf16* Yb = Vt + perBuf;

  qkv_gemm_kernel<<<dim3(N3 / 128, (Bb * Ss) / 64), 256, 0, stream>>>(
      x, W_attn, b_attn, Qb, Kb, Vt);
  attn_kernel<<<dim3(Ss / 128, Hh, Bb), 256, 0, stream>>>(Qb, Kb, Vt, Yb);
  proj_gemm_kernel<<<dim3(Ee / 128, (Bb * Ss) / 64), 256, 0, stream>>>(
      Yb, W_proj, b_proj, (float*)d_out);

  (void)in_sizes; (void)n_in; (void)out_size; (void)ws_size;
}
